// AttentionModel_49108656062795
// MI455X (gfx1250) — matrix operand order repaired and checked by reading, not compiled, not run on hardware
//
#include <hip/hip_runtime.h>
#include <hip/hip_bf16.h>
#include <math.h>

// ---------------- problem constants ----------------
#define BATCH 32
#define NNODE 500
#define DMODEL 256
#define NHEAD 8
#define DK 32
#define FFDIM 512
#define BN (BATCH * NNODE)      // 16000
#define NEGBIG (-1e9f)

typedef __attribute__((ext_vector_type(16))) _Float16 v16h;
typedef __attribute__((ext_vector_type(8)))  float    v8f;
typedef __attribute__((ext_vector_type(4)))  unsigned u32x4;
typedef __attribute__((ext_vector_type(8)))  unsigned u32x8;

// =====================================================================
// init embed: h[b,n,d] = loc0*W[0,d] + loc1*W[1,d] + prize*W[2,d] + b[d]
// =====================================================================
__global__ __launch_bounds__(256) void k_init_embed(
    const float* __restrict__ loc, const float* __restrict__ prize,
    const float* __restrict__ Wi, const float* __restrict__ bi,
    float* __restrict__ h)
{
    size_t i = (size_t)blockIdx.x * blockDim.x + threadIdx.x;
    if (i >= (size_t)BN * DMODEL) return;
    int d = (int)(i & (DMODEL - 1));
    size_t node = i >> 8;  // b*N + n
    h[i] = loc[node * 2 + 0] * Wi[0 * DMODEL + d]
         + loc[node * 2 + 1] * Wi[1 * DMODEL + d]
         + prize[node]       * Wi[2 * DMODEL + d]
         + bi[d];
}

// =====================================================================
// WMMA GEMM: C[M,Nc] = A[M,K] @ W[K,Nc] (+bias)(+residual)(relu)
// One wave per 16x16 output tile; K multiple of 32 (<=512).
// The Kx16 weight tile is DMA'd into LDS by the Tensor Data Mover
// (tensor_load_to_lds, 2-SGPR-group form, tracked with TENSORcnt);
// A rows stream from global with prefetch.
// Fragment layouts used here (wave32; k contiguous per lane half, not ISA table order):
//   A 16x32 f16: lane L -> row L&15; k=(L>>4)*16+j
//   B 32x16 f16: lane L -> col L&15; k=(L>>4)*16+j
//   C 16x16 f32: lane L, vgpr r -> row r+(L>>4)*8, col L&15
// =====================================================================
__global__ __launch_bounds__(32) void k_gemm16(
    const float* __restrict__ A, const float* __restrict__ W,
    const float* __restrict__ bias, const float* __restrict__ residual,
    float* __restrict__ C, int M, int K, int Nc, int relu)
{
    __shared__ float Bt[512 * 16];          // K x 16 weight tile (<=32KB)
    int tiles_n = Nc >> 4;
    int mt = blockIdx.x / tiles_n;
    int nt = blockIdx.x % tiles_n;
    int lane = threadIdx.x & 31;
    int hf   = lane >> 4;      // 0/1
    int l15  = lane & 15;
    int row  = mt * 16 + l15;  // A row for this lane's A-fragment
    const float* Arow = A + (size_t)row * K;

    // ---- TDM: DMA the Kx16 f32 weight tile (row stride Nc) into LDS ----
    {
        unsigned long long gaddr =
            (unsigned long long)(const void*)(W + (size_t)nt * 16);
        unsigned lds_addr = (unsigned)(unsigned long long)(const void*)&Bt[0];
        u32x4 g0;
        g0[0] = 1u;                                   // count=1 valid descriptor
        g0[1] = lds_addr;                             // LDS byte address
        g0[2] = (unsigned)gaddr;                      // global_addr[31:0]
        g0[3] = (unsigned)(gaddr >> 32) | (2u << 30); // global_addr[56:32] | type=2
        u32x8 g1;
        g1[0] = 2u << 16;                 // workgroup_mask=0, data_size=2 (4B)
        g1[1] = 16u << 16;                // tensor_dim0 = 16  (bits 79:48)
        g1[2] = (unsigned)K << 16;        // tensor_dim1 = K   (bits 111:80, lo16)
        g1[3] = 16u << 16;                // tensor_dim1 hi=0 | tile_dim0 = 16
        g1[4] = (unsigned)K & 0xffffu;    // tile_dim1 = K, tile_dim2 = 0
        g1[5] = (unsigned)Nc;             // tensor_dim0_stride[31:0] = Nc
        g1[6] = 0u;                       // stride hi, dim1_stride lo
        g1[7] = 0u;
        asm volatile("tensor_load_to_lds %0, %1" :: "s"(g0), "s"(g1) : "memory");
    }

    v8f acc = {};
    __builtin_amdgcn_s_wait_tensorcnt(0);   // weight tile resident in LDS
    for (int kk = 0; kk < K; kk += 32) {
        if (kk + 32 < K) __builtin_prefetch(Arow + kk + 32, 0, 1);
        v16h a, b;
#pragma unroll
        for (int j = 0; j < 8; ++j) {
            a[j]     = (_Float16)Arow[kk + hf * 16 + j];
            a[j + 8] = (_Float16)Arow[kk + hf * 16 + 8 + j];
        }
#pragma unroll
        for (int j = 0; j < 16; ++j)
            b[j] = (_Float16)Bt[(kk + hf * 16 + j) * 16 + l15];
        acc = __builtin_amdgcn_wmma_f32_16x16x32_f16(
                  false, a, false, b, (short)0, acc, false, false);
    }
#pragma unroll
    for (int r = 0; r < 8; ++r) {
        int m = mt * 16 + r + hf * 8;
        int n = nt * 16 + l15;
        if (m < M && n < Nc) {
            float v = acc[r];
            if (bias)     v += bias[n];
            if (residual) v += residual[(size_t)m * Nc + n];
            if (relu)     v = v > 0.f ? v : 0.f;
            C[(size_t)m * Nc + n] = v;
        }
    }
}

// =====================================================================
// Flash-style MHA: one wave per (b, head, 16-row block).
// qkv: [B,N,3*D] with q|k|v, per-head slice h*32..h*32+31.
// scores (16x512) staged in LDS, softmax, then P@V via WMMA.
// out: [B,N,D] heads merged.
// =====================================================================
__global__ __launch_bounds__(32) void k_attn(
    const float* __restrict__ qkv, float* __restrict__ out)
{
    __shared__ float S[16][512];        // 32 KB
    int blk = blockIdx.x;
    int rb = blk & 31;                  // row block 0..31 (ceil(500/16))
    int h  = (blk >> 5) & (NHEAD - 1);
    int b  = blk >> 8;
    int lane = threadIdx.x;
    int hf = lane >> 4, l15 = lane & 15;
    int rbase = rb * 16;
    const float scale = 0.17677669529663687f;  // 1/sqrt(32)

    // Q fragment (DK=32 == one WMMA K-step), rows rbase..rbase+15
    v16h aq;
    {
        int m = rbase + l15;
        const float* qr = qkv + ((size_t)(b * NNODE + (m < NNODE ? m : NNODE - 1))) * (3 * DMODEL) + h * DK;
        bool valid = m < NNODE;
#pragma unroll
        for (int j = 0; j < 8; ++j) {
            aq[j]     = valid ? (_Float16)qr[hf * 16 + j]      : (_Float16)0.f;
            aq[j + 8] = valid ? (_Float16)qr[hf * 16 + 8 + j] : (_Float16)0.f;
        }
    }
    // phase 1: S = Q K^T * scale  (16 cols per iter)
    for (int c = 0; c < 32; ++c) {
        int n = c * 16 + l15;
        const float* kr = qkv + ((size_t)(b * NNODE + (n < NNODE ? n : NNODE - 1))) * (3 * DMODEL) + DMODEL + h * DK;
        v16h bk;
#pragma unroll
        for (int j = 0; j < 16; ++j)
            bk[j] = (n < NNODE) ? (_Float16)kr[hf * 16 + j] : (_Float16)0.f;
        v8f s = {};
        s = __builtin_amdgcn_wmma_f32_16x16x32_f16(
                false, aq, false, bk, (short)0, s, false, false);
#pragma unroll
        for (int r = 0; r < 8; ++r)
            S[r + hf * 8][c * 16 + l15] = (n < NNODE) ? s[r] * scale : -1e30f;
    }
    __syncthreads();
    // phase 2: row softmax in LDS
    for (int r = 0; r < 16; ++r) {
        float mx = -3.4e38f;
        for (int n = lane; n < 512; n += 32) mx = fmaxf(mx, S[r][n]);
        for (int m = 16; m; m >>= 1) mx = fmaxf(mx, __shfl_xor(mx, m, 32));
        float sm = 0.f;
        for (int n = lane; n < 512; n += 32) {
            float e = __expf(S[r][n] - mx);
            S[r][n] = e; sm += e;
        }
        for (int m = 16; m; m >>= 1) sm += __shfl_xor(sm, m, 32);
        float inv = 1.f / sm;
        for (int n = lane; n < 512; n += 32) S[r][n] *= inv;
    }
    __syncthreads();
    // phase 3: O = P V  (32 nodes/iter; two 16-wide dk halves)
    v8f o0 = {}, o1 = {};
    for (int c = 0; c < 16; ++c) {
        v16h ap;
#pragma unroll
        for (int j = 0; j < 8; ++j) {
            ap[j]     = (_Float16)S[l15][c * 32 + hf * 16 + j];
            ap[j + 8] = (_Float16)S[l15][c * 32 + hf * 16 + 8 + j];
        }
        v16h bv0, bv1;
#pragma unroll
        for (int j = 0; j < 16; ++j) {
            int n = c * 32 + hf * 16 + j;
            const float* vr = qkv + ((size_t)(b * NNODE + (n < NNODE ? n : NNODE - 1))) * (3 * DMODEL) + 2 * DMODEL + h * DK;
            bv0[j] = (_Float16)vr[l15];
            bv1[j] = (_Float16)vr[16 + l15];
        }
        o0 = __builtin_amdgcn_wmma_f32_16x16x32_f16(
                 false, ap, false, bv0, (short)0, o0, false, false);
        o1 = __builtin_amdgcn_wmma_f32_16x16x32_f16(
                 false, ap, false, bv1, (short)0, o1, false, false);
    }
#pragma unroll
    for (int r = 0; r < 8; ++r) {
        int n = rbase + r + hf * 8;
        if (n < NNODE) {
            size_t o = ((size_t)(b * NNODE + n)) * DMODEL + h * DK;
            out[o + l15]      = o0[r];
            out[o + 16 + l15] = o1[r];
        }
    }
}

// =====================================================================
// BatchNorm train-mode stats: one block per feature (256 features)
// mv[0..255]=mean, mv[256..511]=var (biased)
// =====================================================================
__global__ __launch_bounds__(256) void k_bn_stats(const float* __restrict__ x,
                                                  float* __restrict__ mv)
{
    int f = blockIdx.x;
    float s = 0.f, s2 = 0.f;
    for (int i = threadIdx.x; i < BN; i += 256) {
        float v = x[(size_t)i * DMODEL + f];
        s += v; s2 += v * v;
    }
    __shared__ float rs[256], rs2[256];
    rs[threadIdx.x] = s; rs2[threadIdx.x] = s2;
    __syncthreads();
    for (int st = 128; st; st >>= 1) {
        if (threadIdx.x < st) { rs[threadIdx.x] += rs[threadIdx.x + st];
                                rs2[threadIdx.x] += rs2[threadIdx.x + st]; }
        __syncthreads();
    }
    if (threadIdx.x == 0) {
        float m = rs[0] / (float)BN;
        mv[f] = m;
        mv[DMODEL + f] = rs2[0] / (float)BN - m * m;
    }
}

__global__ __launch_bounds__(256) void k_bn_apply(float* __restrict__ x,
    const float* __restrict__ mv, const float* __restrict__ g,
    const float* __restrict__ bb)
{
    size_t i = (size_t)blockIdx.x * blockDim.x + threadIdx.x;
    if (i >= (size_t)BN * DMODEL) return;
    int f = (int)(i & (DMODEL - 1));
    x[i] = g[f] * (x[i] - mv[f]) * rsqrtf(mv[DMODEL + f] + 1e-5f) + bb[f];
}

// mean over nodes: out[b,d] = mean_n emb[b,n,d]
__global__ __launch_bounds__(256) void k_mean_nodes(const float* __restrict__ emb,
                                                    float* __restrict__ o)
{
    int i = blockIdx.x * blockDim.x + threadIdx.x;
    if (i >= BATCH * DMODEL) return;
    int b = i >> 8, d = i & (DMODEL - 1);
    float s = 0.f;
    for (int n = 0; n < NNODE; ++n)
        s += emb[((size_t)(b * NNODE + n)) * DMODEL + d];
    o[i] = s / (float)NNODE;
}

// =====================================================================
// Persistent greedy decode: one 256-thread block (8 waves) per batch.
// proj = emb@Wne: per node 768 floats -> gk | gv | lk.
// out: log_p at [b*N*N + i*N + n], pi (as float) at [B*N*N + b*N + i].
// =====================================================================
__global__ __launch_bounds__(256) void k_decode(
    const float* __restrict__ emb, const float* __restrict__ proj,
    const float* __restrict__ fctx, const float* __restrict__ Wsc,
    const float* __restrict__ Wout, float* __restrict__ out)
{
    int b = blockIdx.x;
    int t = threadIdx.x;
    int lane = t & 31, wv = t >> 5;
    __shared__ float cur[DMODEL], q[DMODEL], glim[DMODEL], fin[DMODEL];
    __shared__ float att[NHEAD][512];
    __shared__ float logits[512];
    __shared__ float red[256];
    __shared__ int   redi[256];
    __shared__ unsigned char vis[512];
    __shared__ int prev_s;
    for (int n = t; n < 512; n += 256) vis[n] = 0;
    if (t == 0) prev_s = 0;
    __syncthreads();

    const float inv_sqrt_dk = 0.17677669529663687f;   // 1/sqrt(32)
    const float inv_sqrt_d  = 0.0625f;                // 1/sqrt(256)

    for (int i = 0; i < NNODE; ++i) {
        int prev = prev_s;
        cur[t] = emb[((size_t)(b * NNODE + prev)) * DMODEL + t];
        __syncthreads();
        float rem = 1.0f - (float)i / (float)NNODE;
        // q = fixed_ctx + [cur, rem] @ Wsc   (Wsc: [257,256])
        {
            float a = fctx[b * DMODEL + t] + rem * Wsc[DMODEL * DMODEL + t];
            for (int j = 0; j < DMODEL; ++j) a += cur[j] * Wsc[j * DMODEL + t];
            q[t] = a;
        }
        __syncthreads();
        // masked glimpse compat
        for (int idx = t; idx < NHEAD * NNODE; idx += 256) {
            int h = idx / NNODE, n = idx % NNODE;
            const float* gk = proj + ((size_t)(b * NNODE + n)) * (3 * DMODEL) + h * DK;
            float s = 0.f;
            for (int d = 0; d < DK; ++d) s += q[h * DK + d] * gk[d];
            att[h][n] = vis[n] ? NEGBIG : s * inv_sqrt_dk;
        }
        __syncthreads();
        // per-head softmax: wave wv handles head wv
        {
            int h = wv;
            float mx = -3.4e38f;
            for (int n = lane; n < NNODE; n += 32) mx = fmaxf(mx, att[h][n]);
            for (int m = 16; m; m >>= 1) mx = fmaxf(mx, __shfl_xor(mx, m, 32));
            float sm = 0.f;
            for (int n = lane; n < NNODE; n += 32) {
                float e = __expf(att[h][n] - mx);
                att[h][n] = e; sm += e;
            }
            for (int m = 16; m; m >>= 1) sm += __shfl_xor(sm, m, 32);
            float inv = 1.f / sm;
            for (int n = lane; n < NNODE; n += 32) att[h][n] *= inv;
        }
        __syncthreads();
        // glimpse: output dim t = h*32+dk
        {
            int h = t >> 5;
            float g = 0.f;
            for (int n = 0; n < NNODE; ++n)
                g += att[h][n] * proj[((size_t)(b * NNODE + n)) * (3 * DMODEL) + DMODEL + t];
            glim[t] = g;
        }
        __syncthreads();
        // fin = glimpse @ Wout
        {
            float s = 0.f;
            for (int j = 0; j < DMODEL; ++j) s += glim[j] * Wout[j * DMODEL + t];
            fin[t] = s;
        }
        __syncthreads();
        // tanh-clipped masked logits + local argmax
        float lmax = -3.4e38f; int lidx = 0;
        for (int n = t; n < NNODE; n += 256) {
            const float* lk = proj + ((size_t)(b * NNODE + n)) * (3 * DMODEL) + 2 * DMODEL;
            float s = 0.f;
            for (int d = 0; d < DMODEL; ++d) s += fin[d] * lk[d];
            s = tanhf(s * inv_sqrt_d) * 10.0f;
            if (vis[n]) s = NEGBIG;
            logits[n] = s;
            if (s > lmax) { lmax = s; lidx = n; }
        }
        red[t] = lmax; redi[t] = lidx;
        __syncthreads();
        for (int st = 128; st; st >>= 1) {
            if (t < st) {
                if (red[t + st] > red[t] ||
                    (red[t + st] == red[t] && redi[t + st] < redi[t])) {
                    red[t] = red[t + st]; redi[t] = redi[t + st];
                }
            }
            __syncthreads();
        }
        float gmax = red[0]; int sel = redi[0];
        __syncthreads();
        float se = 0.f;
        for (int n = t; n < NNODE; n += 256) se += __expf(logits[n] - gmax);
        red[t] = se;
        __syncthreads();
        for (int st = 128; st; st >>= 1) {
            if (t < st) red[t] += red[t + st];
            __syncthreads();
        }
        float lse = __logf(red[0]) + gmax;
        for (int n = t; n < NNODE; n += 256)
            out[(size_t)b * NNODE * NNODE + (size_t)i * NNODE + n] = logits[n] - lse;
        if (t == 0) {
            out[(size_t)BATCH * NNODE * NNODE + b * NNODE + i] = (float)sel;
            vis[sel] = 1;
            prev_s = sel;
        }
        __syncthreads();
    }
}

// =====================================================================
extern "C" void kernel_launch(void* const* d_in, const int* in_sizes, int n_in,
                              void* d_out, int out_size, void* d_ws, size_t ws_size,
                              hipStream_t stream)
{
    (void)in_sizes; (void)n_in; (void)out_size; (void)ws_size;
    const float* loc    = (const float*)d_in[0];
    const float* prize  = (const float*)d_in[1];
    const float* W_init = (const float*)d_in[2];
    const float* b_init = (const float*)d_in[3];
    const float* Wne    = (const float*)d_in[24];
    const float* Wfc    = (const float*)d_in[25];
    const float* Wsc    = (const float*)d_in[26];
    const float* Wout   = (const float*)d_in[27];

    float* ws   = (float*)d_ws;
    float* hA   = ws;                       // [BN,256]
    float* hB   = hA + (size_t)BN * DMODEL; // [BN,256]
    float* tmp1 = hB + (size_t)BN * DMODEL; // [BN,768]
    float* tmp2 = tmp1 + (size_t)BN * 3 * DMODEL; // [BN,256]
    float* mv   = tmp2 + (size_t)BN * DMODEL;     // 512
    float* ctxm = mv + 2 * DMODEL;          // [B,256]
    float* fctx = ctxm + BATCH * DMODEL;    // [B,256]

    const int elemBlocks = (BN * DMODEL + 255) / 256;
    k_init_embed<<<elemBlocks, 256, 0, stream>>>(loc, prize, W_init, b_init, hA);

    for (int l = 0; l < 2; ++l) {
        const float* Wqkv = (const float*)d_in[4 + 10 * l + 0];
        const float* Wo   = (const float*)d_in[4 + 10 * l + 1];
        const float* g1   = (const float*)d_in[4 + 10 * l + 2];
        const float* b1   = (const float*)d_in[4 + 10 * l + 3];
        const float* W1   = (const float*)d_in[4 + 10 * l + 4];
        const float* bf1  = (const float*)d_in[4 + 10 * l + 5];
        const float* W2   = (const float*)d_in[4 + 10 * l + 6];
        const float* bf2  = (const float*)d_in[4 + 10 * l + 7];
        const float* g2   = (const float*)d_in[4 + 10 * l + 8];
        const float* b2   = (const float*)d_in[4 + 10 * l + 9];

        k_gemm16<<<(BN / 16) * (768 / 16), 32, 0, stream>>>(
            hA, Wqkv, nullptr, nullptr, tmp1, BN, DMODEL, 3 * DMODEL, 0);
        k_attn<<<BATCH * NHEAD * 32, 32, 0, stream>>>(tmp1, tmp2);
        k_gemm16<<<(BN / 16) * (256 / 16), 32, 0, stream>>>(
            tmp2, Wo, nullptr, hA, hB, BN, DMODEL, DMODEL, 0);
        k_bn_stats<<<DMODEL, 256, 0, stream>>>(hB, mv);
        k_bn_apply<<<elemBlocks, 256, 0, stream>>>(hB, mv, g1, b1);
        k_gemm16<<<(BN / 16) * (512 / 16), 32, 0, stream>>>(
            hB, W1, bf1, nullptr, tmp1, BN, DMODEL, FFDIM, 1);
        k_gemm16<<<(BN / 16) * (256 / 16), 32, 0, stream>>>(
            tmp1, W2, bf2, hB, hA, BN, FFDIM, DMODEL, 0);
        k_bn_stats<<<DMODEL, 256, 0, stream>>>(hA, mv);
        k_bn_apply<<<elemBlocks, 256, 0, stream>>>(hA, mv, g2, b2);
    }

    k_mean_nodes<<<(BATCH * DMODEL + 255) / 256, 256, 0, stream>>>(hA, ctxm);
    k_gemm16<<<(BATCH / 16) * (256 / 16), 32, 0, stream>>>(
        ctxm, Wfc, nullptr, nullptr, fctx, BATCH, DMODEL, DMODEL, 0);
    k_gemm16<<<(BN / 16) * (768 / 16), 32, 0, stream>>>(
        hA, Wne, nullptr, nullptr, tmp1, BN, DMODEL, 3 * DMODEL, 0);

    k_decode<<<BATCH, 256, 0, stream>>>(hA, tmp1, fctx, Wsc, Wout, (float*)d_out);
}
